// MultiHeadAttention_69106023793216
// MI455X (gfx1250) — compile-verified, matrix operand order checked by reading, not run on hardware
//
#include <hip/hip_runtime.h>
#include <hip/hip_bf16.h>

// ---------------------------------------------------------------------------
// MHA forward for gfx1250 (MI455X), bf16 WMMA + f32 accumulation.
// Round 2: 32x64 register-double-buffered GEMM tiles; flash attention with
// async global->LDS double-buffered K/V tiles (ASYNCcnt) + ds_load fragments.
// ---------------------------------------------------------------------------

typedef __bf16 bf16_t;
typedef __attribute__((ext_vector_type(16))) __bf16 v16bf;
typedef __attribute__((ext_vector_type(8)))  float  v8f;
typedef __attribute__((ext_vector_type(4)))  float  f32x4;

constexpr int SEQ = 2048;   // sequence length
constexpr int DX  = 1024;   // model input dim
constexpr int DK  = 64;     // head dim (d_k == d_v)
constexpr int DM  = 1024;   // output dim
constexpr int NH  = 16;     // heads

__device__ __forceinline__ v8f wmma_bf16(v16bf a, v16bf b, v8f c) {
  // D(16x16,f32) = A(16x32,bf16) * B(32x16,bf16) + C
  return __builtin_amdgcn_wmma_f32_16x16x32_bf16(
      /*neg_a=*/false, a, /*neg_b=*/false, b,
      /*c_mod=*/(short)0, c, /*reuse_a=*/false, /*reuse_b=*/false);
}

union FragU { v16bf v; f32x4 q[2]; };
union Pack8 { bf16_t b[8]; f32x4 q; };

// 16x32 bf16 A-fragment (B-fragment by symmetry from the transposed operand):
// lane L holds row (row0 + L%16); elems 0..7 -> k = k0 + 8*(L/16) + i,
// elems 8..15 -> k = k0 + 16 + 8*(L/16) + i. Two 16B loads per lane.
// Works for global or LDS-backed pointers (addrspace inferred after inline).
__device__ __forceinline__ v16bf load_frag(const bf16_t* __restrict__ base,
                                           int row0, int ld, int k0, int lane) {
  const int row = row0 + (lane & 15);
  const int kk  = k0 + ((lane >> 4) << 3);
  const bf16_t* p = base + (size_t)row * ld + kk;
  FragU u;
  u.q[0] = *reinterpret_cast<const f32x4*>(p);
  u.q[1] = *reinterpret_cast<const f32x4*>(p + 16);
  return u.v;
}

// Async global->LDS 16B copy (per lane). lds_off is the wave-relative LDS
// byte address = low 32 bits of a generic pointer to __shared__ memory.
__device__ __forceinline__ void async_copy_b128(unsigned lds_off,
                                                unsigned long long gaddr) {
  asm volatile("global_load_async_to_lds_b128 %0, %1, off"
               :: "v"(lds_off), "v"(gaddr)
               : "memory");
}

// ---------------------------------------------------------------------------
// Stage 1: f32 -> bf16 conversion / per-head weight transpose
// ---------------------------------------------------------------------------
__global__ void cvt_kernel(const float* __restrict__ src, bf16_t* __restrict__ dst, int n) {
  int i = blockIdx.x * blockDim.x + threadIdx.x;
  if (i < n) dst[i] = (bf16_t)src[i];
}

// W: (NH, DX, DK) f32 row-major  ->  Wt: (NH, DK, DX) bf16 row-major
__global__ void transpose_w_kernel(const float* __restrict__ W, bf16_t* __restrict__ Wt) {
  int i = blockIdx.x * blockDim.x + threadIdx.x;      // over NH*DX*DK = 16*65536
  int h = i >> 16;
  int rem = i & 0xFFFF;
  int x = rem >> 6;        // 0..1023
  int n = rem & 63;        // 0..63
  Wt[((size_t)h << 16) + (size_t)n * DX + x] = (bf16_t)W[i];
}

// ---------------------------------------------------------------------------
// Stage 2: per-head projection GEMM. One wave computes a 32(M) x 64(N) tile,
// register double-buffered over k so the 8 WMMAs of a k-step overlap the
// fragment loads of the next one.
//   X  : (SEQ, DX) bf16        Wt : (NH, DK, DX) bf16 (pre-transposed B)
//   out: (NH, SEQ, DK) bf16 if !transpose_out, else (NH, DK, SEQ) bf16
// ---------------------------------------------------------------------------
__global__ __launch_bounds__(32) void proj_kernel(const bf16_t* __restrict__ X,
                                                  const bf16_t* __restrict__ Wt,
                                                  bf16_t* __restrict__ out,
                                                  int transpose_out) {
  const int lane = threadIdx.x;
  const int m0 = blockIdx.x * 32;
  const int h  = blockIdx.y;
  const bf16_t* Wh = Wt + ((size_t)h * DK * DX);

  v8f zero = {0.f,0.f,0.f,0.f,0.f,0.f,0.f,0.f};
  v8f acc[8] = {zero, zero, zero, zero, zero, zero, zero, zero}; // [mt*4+nt]

  v16bf a0 = load_frag(X, m0,      DX, 0, lane);
  v16bf a1 = load_frag(X, m0 + 16, DX, 0, lane);
  v16bf b0 = load_frag(Wh, 0,  DX, 0, lane);
  v16bf b1 = load_frag(Wh, 16, DX, 0, lane);
  v16bf b2 = load_frag(Wh, 32, DX, 0, lane);
  v16bf b3 = load_frag(Wh, 48, DX, 0, lane);

  for (int k0 = 0; k0 < DX; k0 += 32) {
    const int kn = k0 + 32;
    v16bf na0, na1, nb0, nb1, nb2, nb3;
    if (kn < DX) {
      na0 = load_frag(X, m0,      DX, kn, lane);
      na1 = load_frag(X, m0 + 16, DX, kn, lane);
      nb0 = load_frag(Wh, 0,  DX, kn, lane);
      nb1 = load_frag(Wh, 16, DX, kn, lane);
      nb2 = load_frag(Wh, 32, DX, kn, lane);
      nb3 = load_frag(Wh, 48, DX, kn, lane);
    }
    acc[0] = wmma_bf16(a0, b0, acc[0]);
    acc[1] = wmma_bf16(a0, b1, acc[1]);
    acc[2] = wmma_bf16(a0, b2, acc[2]);
    acc[3] = wmma_bf16(a0, b3, acc[3]);
    acc[4] = wmma_bf16(a1, b0, acc[4]);
    acc[5] = wmma_bf16(a1, b1, acc[5]);
    acc[6] = wmma_bf16(a1, b2, acc[6]);
    acc[7] = wmma_bf16(a1, b3, acc[7]);
    if (kn < DX) {
      a0 = na0; a1 = na1;
      b0 = nb0; b1 = nb1; b2 = nb2; b3 = nb3;
    }
  }

  const int half = lane >> 4, col = lane & 15;
  if (!transpose_out) {
    bf16_t* oh = out + (size_t)h * SEQ * DK;
#pragma unroll
    for (int mt = 0; mt < 2; ++mt)
#pragma unroll
      for (int nt = 0; nt < 4; ++nt)
#pragma unroll
        for (int r = 0; r < 8; ++r)
          oh[(size_t)(m0 + mt * 16 + r + 8 * half) * DK + nt * 16 + col] =
              (bf16_t)acc[mt * 4 + nt][r];
  } else {
    // write D transposed: lane owns column n -> a contiguous run of 8 rows
    bf16_t* oh = out + (size_t)h * DK * SEQ;
#pragma unroll
    for (int mt = 0; mt < 2; ++mt)
#pragma unroll
      for (int nt = 0; nt < 4; ++nt) {
        Pack8 u;
#pragma unroll
        for (int r = 0; r < 8; ++r) u.b[r] = (bf16_t)acc[mt * 4 + nt][r];
        *reinterpret_cast<f32x4*>(oh + (size_t)(nt * 16 + col) * SEQ +
                                  m0 + mt * 16 + 8 * half) = u.q;
      }
  }
}

// ---------------------------------------------------------------------------
// Stage 3: flash attention, one wave per (head, 16-query tile).
// S^T = Kp * Qp^T so each lane owns one query column (in-lane softmax + one
// shfl_xor(16)); two stacked S^T fragments convert IN-LANE into the K=32
// B-fragment of O^T = Vp^T * P^T.
// K/V tiles are double-buffered in LDS via async global->LDS copies:
// prefetch tile tt+32 (16 async B128 ops) while computing tile tt from LDS.
//   Qp,Kp: (NH, SEQ, DK) bf16   Vpt: (NH, DK, SEQ) bf16
//   concat: (SEQ, NH*DK) bf16
// ---------------------------------------------------------------------------
__device__ __forceinline__ void attn_prefetch(const bf16_t* __restrict__ Kh,
                                              const bf16_t* __restrict__ Vh,
                                              int tt, bf16_t* ldsK, bf16_t* ldsV,
                                              int lane) {
  // K tile: rows tt..tt+31, all 64 channels -> one contiguous 4KB block.
  const unsigned kb = (unsigned)(size_t)(void*)ldsK;       // LDS byte offset
  const unsigned long long kg = (unsigned long long)(size_t)(Kh + (size_t)tt * DK);
#pragma unroll
  for (int j = 0; j < 8; ++j) {
    const unsigned c = (unsigned)(j * 32 + lane) * 16u;    // 16B chunk
    async_copy_b128(kb + c, kg + c);
  }
  // V tile: 64 v-rows x 32 keys, row stride SEQ in global, packed ld=32 in LDS.
  const unsigned vb = (unsigned)(size_t)(void*)ldsV;
  const unsigned long long vg = (unsigned long long)(size_t)Vh +
                                (unsigned long long)tt * 2u;
#pragma unroll
  for (int j = 0; j < 8; ++j) {
    const int c = j * 32 + lane;                           // 0..255
    const int row = c >> 2;
    const int o = (c & 3) * 16;
    async_copy_b128(vb + (unsigned)(row * 64 + o),
                    vg + (unsigned long long)row * (SEQ * 2) + (unsigned)o);
  }
}

__global__ __launch_bounds__(32) void attn_kernel(const bf16_t* __restrict__ Qp,
                                                  const bf16_t* __restrict__ Kp,
                                                  const bf16_t* __restrict__ Vpt,
                                                  bf16_t* __restrict__ concat) {
  __shared__ bf16_t ldsK[2][32 * DK];   // 2 x 4KB  (32 keys x 64 ch)
  __shared__ bf16_t ldsV[2][DK * 32];   // 2 x 4KB  (64 v   x 32 keys)

  const int lane = threadIdx.x;
  const int s0 = blockIdx.x * 16;
  const int h  = blockIdx.y;
  const bf16_t* Qh = Qp  + (size_t)h * SEQ * DK;
  const bf16_t* Kh = Kp  + (size_t)h * SEQ * DK;
  const bf16_t* Vh = Vpt + (size_t)h * DK * SEQ;

  // Q as B operand (queries = columns); loaded once from global.
  const v16bf qb0 = load_frag(Qh, s0, DK, 0,  lane);
  const v16bf qb1 = load_frag(Qh, s0, DK, 32, lane);

  v8f zero = {0.f,0.f,0.f,0.f,0.f,0.f,0.f,0.f};
  v8f o0 = zero, o1 = zero, o2 = zero, o3 = zero;   // O^T accum, 64 x 16
  float m_run = -__builtin_inff();
  float l_run = 0.0f;
  const float scale = 0.125f;                        // 1/sqrt(64)

  attn_prefetch(Kh, Vh, 0, ldsK[0], ldsV[0], lane);  // 16 async ops in flight

  int buf = 0;
  for (int tt = 0; tt < SEQ; tt += 32) {
    const int nxt = tt + 32;
    if (nxt < SEQ) {
      attn_prefetch(Kh, Vh, nxt, ldsK[buf ^ 1], ldsV[buf ^ 1], lane);
      // keep the 16 newest (next buffer) in flight; current buffer is done
      asm volatile("s_wait_asynccnt 0x10" ::: "memory");
    } else {
      asm volatile("s_wait_asynccnt 0x0" ::: "memory");
    }

    const bf16_t* kT = ldsK[buf];
    const bf16_t* vT = ldsV[buf];

    // S^T tiles for keys [tt,tt+16) and [tt+16,tt+32): K rows are A operand.
    v16bf ka0 = load_frag(kT, 0,  DK, 0,  lane);
    v16bf ka1 = load_frag(kT, 0,  DK, 32, lane);
    v16bf kc0 = load_frag(kT, 16, DK, 0,  lane);
    v16bf kc1 = load_frag(kT, 16, DK, 32, lane);
    v8f st0 = wmma_bf16(ka0, qb0, zero);
    st0 = wmma_bf16(ka1, qb1, st0);
    v8f st1 = wmma_bf16(kc0, qb0, zero);
    st1 = wmma_bf16(kc1, qb1, st1);

    // online softmax over this lane's query column
    float x0[8], x1[8];
    float vmax = -__builtin_inff();
#pragma unroll
    for (int r = 0; r < 8; ++r) {
      x0[r] = st0[r] * scale;
      x1[r] = st1[r] * scale;
      vmax = fmaxf(vmax, fmaxf(x0[r], x1[r]));
    }
    vmax = fmaxf(vmax, __shfl_xor(vmax, 16, 32));       // combine half-waves
    const float m_new = fmaxf(m_run, vmax);
    const float alpha = __expf(m_run - m_new);          // 0 on first iter
    float p0[8], p1[8], ssum = 0.0f;
#pragma unroll
    for (int r = 0; r < 8; ++r) {
      p0[r] = __expf(x0[r] - m_new);
      p1[r] = __expf(x1[r] - m_new);
      ssum += p0[r] + p1[r];
    }
    ssum += __shfl_xor(ssum, 16, 32);
    l_run = l_run * alpha + ssum;
    m_run = m_new;

    // rescale accumulators
#pragma unroll
    for (int r = 0; r < 8; ++r) {
      o0[r] *= alpha; o1[r] *= alpha; o2[r] *= alpha; o3[r] *= alpha;
    }

    // in-lane pack of P^T into a K=32 B-fragment
    v16bf pb;
#pragma unroll
    for (int r = 0; r < 8; ++r) {
      pb[r]     = (bf16_t)p0[r];
      pb[r + 8] = (bf16_t)p1[r];
    }

    // O^T += Vp^T(tile) * P^T  (4 row tiles cover d_v = 64), V from LDS
    o0 = wmma_bf16(load_frag(vT, 0,  32, 0, lane), pb, o0);
    o1 = wmma_bf16(load_frag(vT, 16, 32, 0, lane), pb, o1);
    o2 = wmma_bf16(load_frag(vT, 32, 32, 0, lane), pb, o2);
    o3 = wmma_bf16(load_frag(vT, 48, 32, 0, lane), pb, o3);

    buf ^= 1;
  }

  const float inv = 1.0f / l_run;
  const int half = lane >> 4, col = lane & 15;
  // lane owns query s0+col; vgpr r -> v = mt*16 + 8*half + r (contiguous!)
  bf16_t* crow = concat + (size_t)(s0 + col) * (NH * DK) + h * DK + 8 * half;
  v8f oo[4] = {o0, o1, o2, o3};
#pragma unroll
  for (int mt = 0; mt < 4; ++mt) {
    Pack8 u;
#pragma unroll
    for (int r = 0; r < 8; ++r) u.b[r] = (bf16_t)(oo[mt][r] * inv);
    *reinterpret_cast<f32x4*>(crow + mt * 16) = u.q;
  }
}

// ---------------------------------------------------------------------------
// Stage 4: out = concat(bf16, SEQ x 1024) @ Wo^T.  Wo row-major is already
// the transposed B operand. One wave -> 32(M) x 64(N) f32 tile, register
// double-buffered like proj_kernel.
// ---------------------------------------------------------------------------
__global__ __launch_bounds__(32) void out_gemm_kernel(const bf16_t* __restrict__ A,
                                                      const bf16_t* __restrict__ Bt,
                                                      float* __restrict__ out) {
  const int lane = threadIdx.x;
  const int m0 = blockIdx.x * 32;
  const int n0 = blockIdx.y * 64;
  constexpr int LDK = NH * DK; // 1024

  v8f zero = {0.f,0.f,0.f,0.f,0.f,0.f,0.f,0.f};
  v8f acc[8] = {zero, zero, zero, zero, zero, zero, zero, zero};

  v16bf a0 = load_frag(A, m0,      LDK, 0, lane);
  v16bf a1 = load_frag(A, m0 + 16, LDK, 0, lane);
  v16bf b0 = load_frag(Bt, n0,      LDK, 0, lane);
  v16bf b1 = load_frag(Bt, n0 + 16, LDK, 0, lane);
  v16bf b2 = load_frag(Bt, n0 + 32, LDK, 0, lane);
  v16bf b3 = load_frag(Bt, n0 + 48, LDK, 0, lane);

  for (int k0 = 0; k0 < LDK; k0 += 32) {
    const int kn = k0 + 32;
    v16bf na0, na1, nb0, nb1, nb2, nb3;
    if (kn < LDK) {
      na0 = load_frag(A, m0,      LDK, kn, lane);
      na1 = load_frag(A, m0 + 16, LDK, kn, lane);
      nb0 = load_frag(Bt, n0,      LDK, kn, lane);
      nb1 = load_frag(Bt, n0 + 16, LDK, kn, lane);
      nb2 = load_frag(Bt, n0 + 32, LDK, kn, lane);
      nb3 = load_frag(Bt, n0 + 48, LDK, kn, lane);
    }
    acc[0] = wmma_bf16(a0, b0, acc[0]);
    acc[1] = wmma_bf16(a0, b1, acc[1]);
    acc[2] = wmma_bf16(a0, b2, acc[2]);
    acc[3] = wmma_bf16(a0, b3, acc[3]);
    acc[4] = wmma_bf16(a1, b0, acc[4]);
    acc[5] = wmma_bf16(a1, b1, acc[5]);
    acc[6] = wmma_bf16(a1, b2, acc[6]);
    acc[7] = wmma_bf16(a1, b3, acc[7]);
    if (kn < LDK) {
      a0 = na0; a1 = na1;
      b0 = nb0; b1 = nb1; b2 = nb2; b3 = nb3;
    }
  }

  const int half = lane >> 4, col = lane & 15;
#pragma unroll
  for (int mt = 0; mt < 2; ++mt)
#pragma unroll
    for (int nt = 0; nt < 4; ++nt)
#pragma unroll
      for (int r = 0; r < 8; ++r)
        out[(size_t)(m0 + mt * 16 + r + 8 * half) * DM + n0 + nt * 16 + col] =
            acc[mt * 4 + nt][r];
}

// ---------------------------------------------------------------------------
extern "C" void kernel_launch(void* const* d_in, const int* in_sizes, int n_in,
                              void* d_out, int out_size, void* d_ws, size_t ws_size,
                              hipStream_t stream) {
  (void)in_sizes; (void)n_in; (void)out_size; (void)ws_size;
  const float* Q  = (const float*)d_in[0];
  const float* K  = (const float*)d_in[1];
  const float* V  = (const float*)d_in[2];
  const float* Wq = (const float*)d_in[3];
  const float* Wk = (const float*)d_in[4];
  const float* Wv = (const float*)d_in[5];
  const float* Wo = (const float*)d_in[6];
  float* out = (float*)d_out;

  char* ws = (char*)d_ws;
  size_t off = 0;
  auto take = [&](size_t bytes) -> char* {
    char* p = ws + off;
    off += (bytes + 255) & ~(size_t)255;
    return p;
  };
  bf16_t* Qbf   = (bf16_t*)take((size_t)SEQ * DX * 2);       // 4 MB
  bf16_t* Kbf   = (bf16_t*)take((size_t)SEQ * DX * 2);       // 4 MB
  bf16_t* Vbf   = (bf16_t*)take((size_t)SEQ * DX * 2);       // 4 MB
  bf16_t* Wqt   = (bf16_t*)take((size_t)NH * DK * DX * 2);   // 2 MB
  bf16_t* Wkt   = (bf16_t*)take((size_t)NH * DK * DX * 2);   // 2 MB
  bf16_t* Wvt   = (bf16_t*)take((size_t)NH * DK * DX * 2);   // 2 MB
  bf16_t* Wobf  = (bf16_t*)take((size_t)DM * NH * DK * 2);   // 2 MB
  bf16_t* Qp    = (bf16_t*)take((size_t)NH * SEQ * DK * 2);  // 4 MB
  bf16_t* Kp    = (bf16_t*)take((size_t)NH * SEQ * DK * 2);  // 4 MB
  bf16_t* Vpt   = (bf16_t*)take((size_t)NH * DK * SEQ * 2);  // 4 MB
  bf16_t* Ccat  = (bf16_t*)take((size_t)SEQ * NH * DK * 2);  // 4 MB
  // total ~36 MB of workspace

  // Stage 1: conversions and weight transposes
  const int nX = SEQ * DX;                 // 2,097,152
  cvt_kernel<<<nX / 256, 256, 0, stream>>>(Q, Qbf, nX);
  cvt_kernel<<<nX / 256, 256, 0, stream>>>(K, Kbf, nX);
  cvt_kernel<<<nX / 256, 256, 0, stream>>>(V, Vbf, nX);
  const int nWo = DM * NH * DK;            // 1,048,576
  cvt_kernel<<<nWo / 256, 256, 0, stream>>>(Wo, Wobf, nWo);
  const int nW = NH * DX * DK;             // 1,048,576
  transpose_w_kernel<<<nW / 256, 256, 0, stream>>>(Wq, Wqt);
  transpose_w_kernel<<<nW / 256, 256, 0, stream>>>(Wk, Wkt);
  transpose_w_kernel<<<nW / 256, 256, 0, stream>>>(Wv, Wvt);

  // Stage 2: projections (one wave per 32x64 tile)
  dim3 pgrid(SEQ / 32, NH);
  proj_kernel<<<pgrid, 32, 0, stream>>>(Qbf, Wqt, Qp, 0);
  proj_kernel<<<pgrid, 32, 0, stream>>>(Kbf, Wkt, Kp, 0);
  proj_kernel<<<pgrid, 32, 0, stream>>>(Vbf, Wvt, Vpt, 1);   // transposed store

  // Stage 3: attention (async LDS double-buffered K/V)
  attn_kernel<<<dim3(SEQ / 16, NH), 32, 0, stream>>>(Qp, Kp, Vpt, Ccat);

  // Stage 4: output projection
  out_gemm_kernel<<<dim3(SEQ / 32, DM / 64), 32, 0, stream>>>(Ccat, Wobf, out);
}
